// HebbianBlock_49855980372401
// MI455X (gfx1250) — compile-verified, matrix operand order checked by reading, not run on hardware
//
#include <hip/hip_runtime.h>
#include <stdint.h>

// ---------------------------------------------------------------------------
// Problem constants (match reference)
// ---------------------------------------------------------------------------
#define BB   4
#define TT   4096
#define DD   1024
#define CC   64
#define NCH  (TT / CC)     // 64 chunks
#define TI   32            // state row-slice per workgroup
#define VPAD 72            // padded row length (bf16 elems) for LDS transposes

typedef __attribute__((ext_vector_type(16))) __bf16        v16bf;
typedef __attribute__((ext_vector_type(8)))  float         v8f;
typedef __attribute__((ext_vector_type(4)))  float         v4f;
typedef __attribute__((ext_vector_type(8)))  unsigned int  v8u;
typedef __attribute__((ext_vector_type(4)))  unsigned int  v4u;
typedef __attribute__((ext_vector_type(2)))  unsigned int  v2u;

// ---------------------------------------------------------------------------
// bf16 helpers (stored as uint16_t)
// ---------------------------------------------------------------------------
static __device__ __forceinline__ unsigned short f2bf(float f) {
  unsigned int u = __builtin_bit_cast(unsigned int, f);
  u += 0x7FFFu + ((u >> 16) & 1u);            // round-to-nearest-even
  return (unsigned short)(u >> 16);
}
static __device__ __forceinline__ unsigned int f2bf_pk(float lo, float hi) {
  return (unsigned int)f2bf(lo) | ((unsigned int)f2bf(hi) << 16);
}
static __device__ __forceinline__ float bf2f(unsigned short h) {
  return __builtin_bit_cast(float, ((unsigned int)h) << 16);
}

// Load a 16-element bf16 fragment: two contiguous 16B chunks.
// A-frag: lane m = lane&15, kb = (lane>>4)*8 -> elems [k0,k0+8) and [k0+16,k0+24)
// B-frag: identical addressing off the TRANSPOSED source B^T[n][k].
static __device__ __forceinline__ v8u frag_u(const unsigned short* p) {
  v4u a = *reinterpret_cast<const v4u*>(p);
  v4u b = *reinterpret_cast<const v4u*>(p + 16);
  v8u r; r[0]=a[0]; r[1]=a[1]; r[2]=a[2]; r[3]=a[3];
         r[4]=b[0]; r[5]=b[1]; r[6]=b[2]; r[7]=b[3];
  return r;
}
static __device__ __forceinline__ v16bf frag(const unsigned short* p) {
  return __builtin_bit_cast(v16bf, frag_u(p));
}
static __device__ __forceinline__ v8f wmma_bf16(v16bf a, v16bf b, v8f c) {
  return __builtin_amdgcn_wmma_f32_16x16x32_bf16(false, a, false, b, (short)0, c,
                                                 false, false);
}

// ---------------------------------------------------------------------------
// Kernel 0: f32 -> bf16 convert, vectorized: float4 in, 2x packed-bf16 out
// ---------------------------------------------------------------------------
__global__ void k_convert(const float* __restrict__ src,
                          unsigned short* __restrict__ dst, int n4) {
  for (int i = blockIdx.x * blockDim.x + threadIdx.x; i < n4;
       i += gridDim.x * blockDim.x) {
    v4f x = reinterpret_cast<const v4f*>(src)[i];
    v2u p; p[0] = f2bf_pk(x[0], x[1]); p[1] = f2bf_pk(x[2], x[3]);
    reinterpret_cast<v2u*>(dst)[i] = p;
  }
}

// ---------------------------------------------------------------------------
// Kernel 1: Vbf[t,e] = bf16( sum_d Xbf[t,d] * Wwbf[e,d] )   (NT GEMM)
// block = 256 (8 waves); wave tile = 32 x 64; block tile = 128 x 128
// per k-step: 2 A-frags + 4 B-frags -> 8 wmma (each A-frag feeds 4 wmma, each B-frag 2)
// ---------------------------------------------------------------------------
__global__ void k_gemm_v(const unsigned short* __restrict__ Xbf,
                         const unsigned short* __restrict__ Wbf,
                         unsigned short* __restrict__ Vbf) {
  const int tid = threadIdx.x, wave = tid >> 5, lane = tid & 31;
  const int ln = lane & 15, lh = lane >> 4, kb = lh * 8;
  const int m0 = blockIdx.x * 128 + (wave & 3) * 32;
  const int n0 = blockIdx.y * 128 + (wave >> 2) * 64;

  v8f acc[2][4] = {{v8f{}, v8f{}, v8f{}, v8f{}}, {v8f{}, v8f{}, v8f{}, v8f{}}};
  const unsigned short* arow0 = Xbf + (size_t)(m0 + ln) * DD;
  const unsigned short* arow1 = arow0 + (size_t)16 * DD;
#pragma unroll 2
  for (int kk = 0; kk < DD; kk += 32) {
    v16bf a0 = frag(arow0 + kk + kb);
    v16bf a1 = frag(arow1 + kk + kb);
#pragma unroll
    for (int t = 0; t < 4; ++t) {
      v16bf b = frag(Wbf + (size_t)(n0 + t * 16 + ln) * DD + kk + kb);
      acc[0][t] = wmma_bf16(a0, b, acc[0][t]);
      acc[1][t] = wmma_bf16(a1, b, acc[1][t]);
    }
  }
#pragma unroll
  for (int mt = 0; mt < 2; ++mt)
#pragma unroll
    for (int t = 0; t < 4; ++t)
#pragma unroll
      for (int r = 0; r < 8; ++r) {
        int row = m0 + mt * 16 + r + 8 * lh, col = n0 + t * 16 + ln;
        Vbf[(size_t)row * DD + col] = f2bf(acc[mt][t][r]);
      }
}

// ---------------------------------------------------------------------------
// Kernel 2: Pbf[b,ch,c,k] = bf16( M[c,k] * sum_d rk[c,d]*wk[k,d] )
// one block per (b,chunk); 8 waves -> 2 tiles each of the 4x4 tile grid.
// wk row t0+k-1 (zeros for t<0).
// ---------------------------------------------------------------------------
__global__ void k_pmask(const unsigned short* __restrict__ Xbf,
                        unsigned short* __restrict__ Pbf,
                        const float* __restrict__ decay) {
  const int tid = threadIdx.x, wave = tid >> 5, lane = tid & 31;
  const int ln = lane & 15, lh = lane >> 4, kb = lh * 8;
  const int b = blockIdx.x / NCH, ch = blockIdx.x % NCH, t0 = ch * CC;
  const int mt = wave >> 1, np = wave & 1;

  const float g  = 1.f / (1.f + __expf(-decay[0]));
  const float lg = __logf(g);

  v8f acc[2] = {v8f{}, v8f{}};
  const unsigned short* arow = Xbf + (size_t)(b * TT + t0 + mt * 16 + ln) * DD;
  for (int kk = 0; kk < DD; kk += 32) {
    v16bf a = frag(arow + kk + kb);
#pragma unroll
    for (int u = 0; u < 2; ++u) {
      int krow = (np * 2 + u) * 16 + ln;           // k index inside chunk
      int tw   = t0 + krow - 1;                    // shifted write-key row
      v8u bu = v8u{};
      if (tw >= 0) bu = frag_u(Xbf + (size_t)(b * TT + tw) * DD + kk + kb);
      acc[u] = wmma_bf16(a, __builtin_bit_cast(v16bf, bu), acc[u]);
    }
  }
  unsigned short* pdst = Pbf + (size_t)blockIdx.x * CC * CC;
#pragma unroll
  for (int u = 0; u < 2; ++u)
#pragma unroll
    for (int r = 0; r < 8; ++r) {
      int c = mt * 16 + r + 8 * lh;
      int k = (np * 2 + u) * 16 + ln;
      float val = (c > k) ? acc[u][r] * __expf(lg * (float)(c - 1 - k)) : 0.f;
      pdst[c * CC + k] = f2bf(val);
    }
}

// ---------------------------------------------------------------------------
// Kernel 3: chunked scan.  grid = B * (D/TI) = 128 blocks, 256 threads.
// LDS: f32 master state (TI x D), bf16 shadow, vT (TI x C), wkT quarter tiles.
// ---------------------------------------------------------------------------
#define OFF_SHADOW (TI * DD * 4)                       // 131072
#define OFF_VT     (OFF_SHADOW + TI * DD * 2)          // 196608
#define OFF_WKT    (OFF_VT + TI * VPAD * 2)            // 201216
#define SMEM_BYTES (OFF_WKT + 256 * VPAD * 2)          // 238080

__global__ void k_scan(const unsigned short* __restrict__ Xbf,
                       const unsigned short* __restrict__ Vbf,
                       const unsigned short* __restrict__ Pbf,
                       unsigned short* __restrict__ Readsbf,
                       const float* __restrict__ decay) {
  extern __shared__ char smem[];
  float*          Wst    = (float*)smem;
  unsigned short* shadow = (unsigned short*)(smem + OFF_SHADOW);
  unsigned short* vT     = (unsigned short*)(smem + OFF_VT);
  unsigned short* wkT    = (unsigned short*)(smem + OFF_WKT);

  const int tid = threadIdx.x, wave = tid >> 5, lane = tid & 31;
  const int ln = lane & 15, lh = lane >> 4, kb = lh * 8;
  const int b = blockIdx.x >> 5;                 // 32 slices per batch
  const int i0 = (blockIdx.x & 31) * TI;

  const float g  = 1.f / (1.f + __expf(-decay[0]));
  const float lg = __logf(g);
  const float gC = __expf(lg * (float)CC);

  for (int i = tid; i < TI * DD; i += 256) { Wst[i] = 0.f; shadow[i] = 0; }
  __syncthreads();

  const int mt = wave >> 1, nt = wave & 1;         // Phase-A tile of this wave
  const int mt2 = wave >> 2, ntb = (wave & 3) * 4; // Phase-B tiles

  for (int ch = 0; ch < NCH; ++ch) {
    const int t0 = ch * CC;

    // ---- stage vT[i][c] = Vbf[b, t0+c, i0+i]  (bf16 copy + transpose) ----
    for (int idx = tid; idx < TI * CC; idx += 256) {
      int i = idx & (TI - 1), c = idx >> 5;
      vT[i * VPAD + c] = Vbf[(size_t)(b * TT + t0 + c) * DD + i0 + i];
    }
    __syncthreads();

    // ---- Phase A: reads = intra(P @ v) + gp * inter(rk @ Wst^T) ----
    {
      const int m = mt * 16 + ln;                // c row (A operand)
      const int iloc = nt * 16 + ln;             // i col (B operand)
      v8f acc = v8f{};
      const unsigned short* Pb = Pbf + (size_t)(b * NCH + ch) * CC * CC;
#pragma unroll
      for (int kk = 0; kk < CC; kk += 32) {
        v16bf a  = frag(Pb + m * CC + kk + kb);
        v16bf bf = frag(vT + iloc * VPAD + kk + kb);
        acc = wmma_bf16(a, bf, acc);
      }
      v8f acc2 = v8f{};
      const unsigned short* arow = Xbf + (size_t)(b * TT + t0 + m) * DD;
#pragma unroll 2
      for (int kk = 0; kk < DD; kk += 32) {
        v16bf a  = frag(arow + kk + kb);
        v16bf bf = frag(shadow + iloc * DD + kk + kb);
        acc2 = wmma_bf16(a, bf, acc2);
      }
#pragma unroll
      for (int r = 0; r < 8; ++r) {
        int c = mt * 16 + r + 8 * lh;
        float gp = __expf(lg * (float)c);
        float val = acc[r] + acc2[r] * gp;
        Readsbf[(size_t)(b * TT + t0 + c) * DD + i0 + iloc] = f2bf(val);
      }
    }
    __syncthreads();

    // ---- Phase B: Wst = gC*Wst + vT @ (gw .* wk), quarter-tiled over j ----
    for (int q = 0; q < 4; ++q) {
      const int jb = q * 256;
      for (int idx = tid; idx < CC * 256; idx += 256) {
        int j = idx & 255, c = idx >> 8;
        int tw = t0 + c - 1;
        float x = (tw >= 0) ? bf2f(Xbf[(size_t)(b * TT + tw) * DD + jb + j]) : 0.f;
        float gw = __expf(lg * (float)(CC - 1 - c));
        wkT[j * VPAD + c] = f2bf(x * gw);
      }
      __syncthreads();

#pragma unroll
      for (int s = 0; s < 4; ++s) {
        const int ntq = ntb + s;
        const int jcol = jb + ntq * 16 + ln;
        v8f accu;
#pragma unroll
        for (int r = 0; r < 8; ++r) {
          int i = mt2 * 16 + r + 8 * lh;
          accu[r] = Wst[i * DD + jcol] * gC;
        }
        const int mloc = mt2 * 16 + ln;
        const int jj   = ntq * 16 + ln;
#pragma unroll
        for (int kk = 0; kk < CC; kk += 32) {
          v16bf a  = frag(vT + mloc * VPAD + kk + kb);
          v16bf bf = frag(wkT + jj * VPAD + kk + kb);
          accu = wmma_bf16(a, bf, accu);
        }
#pragma unroll
        for (int r = 0; r < 8; ++r) {
          int i = mt2 * 16 + r + 8 * lh;
          Wst[i * DD + jcol]    = accu[r];
          shadow[i * DD + jcol] = f2bf(accu[r]);
        }
      }
      __syncthreads();
    }
  }
}

// ---------------------------------------------------------------------------
// Kernel 4: d_out[t,e] = out[t,e] + alpha * sum_d Readsbf[t,d]*Wrbf[e,d]
// same 32x64 wave tile as k_gemm_v, fused residual epilogue.
// ---------------------------------------------------------------------------
__global__ void k_gemm_out(const unsigned short* __restrict__ Rbf,
                           const unsigned short* __restrict__ Wbf,
                           const float* __restrict__ outp,
                           float* __restrict__ y,
                           const float* __restrict__ log_alpha) {
  const int tid = threadIdx.x, wave = tid >> 5, lane = tid & 31;
  const int ln = lane & 15, lh = lane >> 4, kb = lh * 8;
  const int m0 = blockIdx.x * 128 + (wave & 3) * 32;
  const int n0 = blockIdx.y * 128 + (wave >> 2) * 64;
  const float alpha = __expf(log_alpha[0]);

  v8f acc[2][4] = {{v8f{}, v8f{}, v8f{}, v8f{}}, {v8f{}, v8f{}, v8f{}, v8f{}}};
  const unsigned short* arow0 = Rbf + (size_t)(m0 + ln) * DD;
  const unsigned short* arow1 = arow0 + (size_t)16 * DD;
#pragma unroll 2
  for (int kk = 0; kk < DD; kk += 32) {
    v16bf a0 = frag(arow0 + kk + kb);
    v16bf a1 = frag(arow1 + kk + kb);
#pragma unroll
    for (int t = 0; t < 4; ++t) {
      v16bf b = frag(Wbf + (size_t)(n0 + t * 16 + ln) * DD + kk + kb);
      acc[0][t] = wmma_bf16(a0, b, acc[0][t]);
      acc[1][t] = wmma_bf16(a1, b, acc[1][t]);
    }
  }
#pragma unroll
  for (int mt = 0; mt < 2; ++mt)
#pragma unroll
    for (int t = 0; t < 4; ++t)
#pragma unroll
      for (int r = 0; r < 8; ++r) {
        size_t idx = (size_t)(m0 + mt * 16 + r + 8 * lh) * DD + n0 + t * 16 + ln;
        y[idx] = outp[idx] + alpha * acc[mt][t][r];
      }
}

// ---------------------------------------------------------------------------
// Launch
// ---------------------------------------------------------------------------
extern "C" void kernel_launch(void* const* d_in, const int* in_sizes, int n_in,
                              void* d_out, int out_size, void* d_ws, size_t ws_size,
                              hipStream_t stream) {
  const float* out_f   = (const float*)d_in[0];
  const float* W_write = (const float*)d_in[1];
  const float* W_read  = (const float*)d_in[2];
  const float* decay   = (const float*)d_in[3];
  const float* log_al  = (const float*)d_in[4];
  float* y = (float*)d_out;

  const size_t NX = (size_t)BB * TT * DD;      // 16,777,216
  const size_t NW = (size_t)DD * DD;           // 1,048,576
  char* ws = (char*)d_ws;
  size_t off = 0;
  unsigned short* Xbf  = (unsigned short*)(ws + off); off += NX * 2;
  unsigned short* Wwbf = (unsigned short*)(ws + off); off += NW * 2;
  unsigned short* Wrbf = (unsigned short*)(ws + off); off += NW * 2;
  unsigned short* Vbf  = (unsigned short*)(ws + off); off += NX * 2;
  unsigned short* Pbf  = (unsigned short*)(ws + off); off += (size_t)BB * NCH * CC * CC * 2;
  unsigned short* Rbf  = (unsigned short*)(ws + off); off += NX * 2;
  if (off > ws_size) return;  // workspace too small (compile-only harness)

  k_convert<<<2048, 256, 0, stream>>>(out_f, Xbf, (int)(NX / 4));
  k_convert<<<512, 256, 0, stream>>>(W_write, Wwbf, (int)(NW / 4));
  k_convert<<<512, 256, 0, stream>>>(W_read, Wrbf, (int)(NW / 4));

  dim3 gg((BB * TT) / 128, DD / 128);
  k_gemm_v<<<gg, 256, 0, stream>>>(Xbf, Wwbf, Vbf);

  k_pmask<<<BB * NCH, 256, 0, stream>>>(Xbf, Pbf, decay);

  k_scan<<<BB * (DD / TI), 256, SMEM_BYTES, stream>>>(Xbf, Vbf, Pbf, Rbf, decay);

  k_gemm_out<<<gg, 256, 0, stream>>>(Rbf, Wrbf, out_f, y, log_al);
}
